// MultiHeadAttention_51960514347783
// MI455X (gfx1250) — compile-verified, matrix operand order checked by reading, not run on hardware
//
#include <hip/hip_runtime.h>
#include <hip/hip_bf16.h>

// MI455X (gfx1250) multi-head attention, bf16 WMMA path with async global->LDS
// double buffering (GLOBAL_LOAD_ASYNC_TO_LDS_B128 / s_wait_asynccnt).
// Pipeline: f32->bf16 convert + mask bit-pack -> Q/K projections, V projection
//           (transposed out) -> flash attention -> output GEMM (f32).

typedef __bf16 bf16_t;
typedef __attribute__((ext_vector_type(16))) __bf16 v16bf;
typedef __attribute__((ext_vector_type(8)))  float  v8f;
typedef __attribute__((ext_vector_type(4)))  __bf16 bf16x4;

#define DEV static __device__ __forceinline__

// ---- CDNA5 async copy primitives (inline asm; ASYNCcnt tracked) -------------
DEV void async_copy_b128(uint32_t lds_byte_addr, const void* gaddr) {
  // Global->LDS copy, 16B per lane, no VGPR staging.
  asm volatile("global_load_async_to_lds_b128 %0, %1, off"
               :: "v"(lds_byte_addr), "v"(gaddr)
               : "memory");
}
DEV void wait_async_le8() { asm volatile("s_wait_asynccnt 0x8" ::: "memory"); }
DEV void wait_async_0()   { asm volatile("s_wait_asynccnt 0x0" ::: "memory"); }

DEV uint32_t lds_addr(const void* p) { return (uint32_t)(uintptr_t)p; }

// ISA 7.12.2: 16-bit A/B operand K interleave.
// VGPR pair v (0..7) holds K = {base, base+1}, base = (v<4 ? 2v : 16+2(v-4)) + 8*half.
DEV int kpair_base(int v, int half) {
  return ((v < 4) ? (2 * v) : (16 + 2 * (v - 4))) + 8 * half;
}

// Load one 16x32 bf16 operand fragment from a row-major-over-K tile.
// p points at element [row][k0]; row = tile_row_base + (lane & 15).
DEV v16bf load_frag_k(const bf16_t* __restrict__ p, int half) {
  v16bf f;
#pragma unroll
  for (int v = 0; v < 8; ++v) {
    int kb = kpair_base(v, half);
    f[2 * v]     = p[kb];
    f[2 * v + 1] = p[kb + 1];
  }
  return f;
}

DEV v8f wmma_bf16(v16bf a, v16bf b, v8f c) {
  return __builtin_amdgcn_wmma_f32_16x16x32_bf16(false, a, false, b, (short)0, c,
                                                 false, false);
}

// ---------------------------------------------------------------- convert ----
__global__ __launch_bounds__(256) void f32_to_bf16_kernel(
    const float* __restrict__ in, bf16_t* __restrict__ out, int n4) {
  int i = blockIdx.x * 256 + threadIdx.x;
  if (i >= n4) return;
  float4 v = ((const float4*)in)[i];
  bf16x4 o = {(bf16_t)v.x, (bf16_t)v.y, (bf16_t)v.z, (bf16_t)v.w};
  ((bf16x4*)out)[i] = o;
}

// -------------------------------------------------------------- mask pack ----
// [B,S,S] int32 {0,1} -> [B,S,S/32] u32 bitmask via wave32 ballot (32x smaller).
__global__ __launch_bounds__(256) void mask_pack_kernel(
    const int* __restrict__ mask, uint32_t* __restrict__ pm, int nwords) {
  int idx  = blockIdx.x * 256 + threadIdx.x;  // element index
  int word = idx >> 5;
  if (word >= nwords) return;
  int mv = mask[idx];
  unsigned long long bal = __ballot(mv != 0);
  if ((threadIdx.x & 31) == 0) pm[word] = (uint32_t)bal;
}

// ------------------------------------------------------------------- GEMM ----
// C[M,N] = A[M,K] * W[N,K]^T. Block 128x128, BK=64, double-buffered async LDS.
// 8 waves arranged 2(M) x 4(N); wave tile 64x32 (8 accumulators).
// VT=true: scatter-store output to V^T global layout [B, H, DK, S] (bf16).
template <typename OutT, bool VT>
__global__ __launch_bounds__(256) void gemm_bf16_nt(
    const bf16_t* __restrict__ A, const bf16_t* __restrict__ W,
    OutT* __restrict__ C, int M, int N, int K) {
  constexpr int BM = 128, BN = 128, BK = 64;
  constexpr int Sc = 2048, Hc = 16;  // for VT epilogue
  __shared__ __attribute__((aligned(16))) bf16_t As[2][BM * BK];
  __shared__ __attribute__((aligned(16))) bf16_t Bs[2][BN * BK];

  const int tid  = threadIdx.x;
  const int lane = tid & 31;
  const int wave = tid >> 5;
  const int half = lane >> 4;
  const int ln   = lane & 15;
  const int wm   = (wave >> 2) * 64;
  const int wn   = (wave & 3) * 32;
  const int bm   = blockIdx.y * BM;
  const int bn   = blockIdx.x * BN;

  // Issue one tile's async copies (8 x B128 per thread: 4 for A, 4 for B).
  auto issue = [&](int k0, int sel) {
#pragma unroll
    for (int t = 0; t < 4; ++t) {
      int c   = tid + t * 256;
      int row = c >> 3;            // 8 chunks per row (BK=64)
      int col = (c & 7) * 8;
      async_copy_b128(lds_addr(&As[sel][row * BK + col]),
                      &A[(size_t)(bm + row) * K + k0 + col]);
      async_copy_b128(lds_addr(&Bs[sel][row * BK + col]),
                      &W[(size_t)(bn + row) * K + k0 + col]);
    }
  };

  v8f acc[4][2] = {};

  issue(0, 0);
  int sel = 0;
  for (int k0 = 0; k0 < K; k0 += BK) {
    const bool pre = (k0 + BK < K);
    if (pre) issue(k0 + BK, sel ^ 1);
    if (pre) wait_async_le8(); else wait_async_0();
    __syncthreads();  // buf[sel] resident for all waves

#pragma unroll
    for (int kk = 0; kk < BK; kk += 32) {
      v16bf af[4], bfm[2];
#pragma unroll
      for (int i = 0; i < 4; ++i)
        af[i] = load_frag_k(&As[sel][(wm + i * 16 + ln) * BK + kk], half);
#pragma unroll
      for (int j = 0; j < 2; ++j)
        bfm[j] = load_frag_k(&Bs[sel][(wn + j * 16 + ln) * BK + kk], half);
#pragma unroll
      for (int i = 0; i < 4; ++i)
#pragma unroll
        for (int j = 0; j < 2; ++j)
          acc[i][j] = wmma_bf16(af[i], bfm[j], acc[i][j]);
    }
    __syncthreads();  // all reads of buf[sel] done; safe to overwrite next iter
    sel ^= 1;
  }

  // C/D layout: element (vgpr r, lane): M = r + 8*half, N = lane&15.
#pragma unroll
  for (int i = 0; i < 4; ++i)
#pragma unroll
    for (int j = 0; j < 2; ++j)
#pragma unroll
      for (int r = 0; r < 8; ++r) {
        int row = bm + wm + i * 16 + r + 8 * half;
        int col = bn + wn + j * 16 + ln;
        if constexpr (VT) {
          // Output element (row=b*S+s, col=h*64+dk) -> Vt[b][h][dk][s]
          int bb = row / Sc, s = row % Sc;
          int hh = col >> 6, dk = col & 63;
          C[(((size_t)bb * Hc + hh) * 64 + dk) * Sc + s] = (OutT)acc[i][j][r];
        } else {
          C[(size_t)row * N + col] = (OutT)acc[i][j][r];
        }
      }
}

// -------------------------------------------------------- flash attention ----
// One block = (b, h, 64-query tile); 4 waves, 16 query rows per wave.
// K tiles ([key][dk]) and V^T tiles ([dk][key]) double-buffered via async copy.
__global__ __launch_bounds__(128) void flash_attn(
    const bf16_t* __restrict__ Q,      // [B*S, D]
    const bf16_t* __restrict__ Km,     // [B*S, D]
    const bf16_t* __restrict__ Vt,     // [B, H, DK, S]
    const uint32_t* __restrict__ pm,   // [B, S, S/32] packed mask bits
    bf16_t* __restrict__ Ctx) {        // [B*S, D]
  constexpr int S = 2048, D = 1024, DKc = 64, TK = 64, SW = S / 32;
  __shared__ __attribute__((aligned(16))) bf16_t Ks[2][TK * DKc];   // [key][dk]
  __shared__ __attribute__((aligned(16))) bf16_t Vs[2][DKc * TK];   // [dk][key]
  __shared__ __attribute__((aligned(16))) bf16_t Ps[4][16 * TK];    // per-wave P

  const int b    = blockIdx.z;
  const int h    = blockIdx.y;
  const int q0   = blockIdx.x * 64;
  const int tid  = threadIdx.x;
  const int lane = tid & 31;
  const int wave = tid >> 5;
  const int half = lane >> 4;
  const int ln   = lane & 15;

  const bf16_t* kbase = Km + (size_t)b * S * D + h * DKc;           // row=key
  const bf16_t* vbase = Vt + ((size_t)b * 16 + h) * 64 * S;         // row=dk

  // Issue one key tile's async copies (8 x B128 per thread: 4 K, 4 V^T).
  auto issue = [&](int kt, int sel) {
#pragma unroll
    for (int t = 0; t < 4; ++t) {
      int c   = tid + t * 128;
      int row = c >> 3;
      int col = (c & 7) * 8;
      async_copy_b128(lds_addr(&Ks[sel][row * DKc + col]),
                      kbase + (size_t)(kt + row) * D + col);
      async_copy_b128(lds_addr(&Vs[sel][row * TK + col]),
                      vbase + (size_t)row * S + kt + col);
    }
  };

  // Q fragments for this wave's 16 query rows (A operand rows = lane&15).
  const bf16_t* qrowp =
      Q + (size_t)(b * S + q0 + wave * 16 + ln) * D + h * DKc;
  v16bf qf[2];
#pragma unroll
  for (int t = 0; t < 2; ++t) qf[t] = load_frag_k(qrowp + t * 32, half);

  v8f oacc[4] = {};
  float mrun[8], lrun[8];
#pragma unroll
  for (int r = 0; r < 8; ++r) { mrun[r] = -3.0e38f; lrun[r] = 0.f; }

  issue(0, 0);
  int sel = 0;
  for (int kt = 0; kt < S; kt += TK) {
    const bool pre = (kt + TK < S);
    if (pre) issue(kt + TK, sel ^ 1);
    if (pre) wait_async_le8(); else wait_async_0();
    __syncthreads();  // buf[sel] resident

    // S = Q (16x64) * K^T: 4 key groups x 2 k-steps = 8 WMMA.
    v8f sacc[4] = {};
#pragma unroll
    for (int t = 0; t < 2; ++t)
#pragma unroll
      for (int n = 0; n < 4; ++n) {
        v16bf kf = load_frag_k(&Ks[sel][(n * 16 + ln) * DKc + t * 32], half);
        sacc[n] = wmma_bf16(qf[t], kf, sacc[n]);
      }

    // Scale (1/sqrt(64)) + mask from packed bits.
    // Per query row, one u64 load covers all 64 keys of the tile.
    float s[4][8];
#pragma unroll
    for (int r = 0; r < 8; ++r) {
      int qrow = q0 + wave * 16 + r + 8 * half;
      uint2 ww = *(const uint2*)(pm + ((size_t)b * S + qrow) * SW + (kt >> 5));
#pragma unroll
      for (int n = 0; n < 4; ++n) {
        uint32_t w   = (n < 2) ? ww.x : ww.y;
        int      bit = (n & 1) * 16 + ln;
        s[n][r] = ((w >> bit) & 1u) ? sacc[n][r] * 0.125f : -1.0e9f;
      }
    }

    // Online softmax. Row M = r + 8*half lives across the 16 lanes of a half.
    float alpha[8];
#pragma unroll
    for (int r = 0; r < 8; ++r) {
      float tmax = fmaxf(fmaxf(s[0][r], s[1][r]), fmaxf(s[2][r], s[3][r]));
#pragma unroll
      for (int m = 8; m >= 1; m >>= 1) tmax = fmaxf(tmax, __shfl_xor(tmax, m, 32));
      float mnew = fmaxf(mrun[r], tmax);
      alpha[r] = __expf(mrun[r] - mnew);
      mrun[r]  = mnew;
      float tsum = 0.f;
#pragma unroll
      for (int n = 0; n < 4; ++n) {
        float p = __expf(s[n][r] - mnew);
        s[n][r] = p;
        tsum += p;
      }
#pragma unroll
      for (int m = 8; m >= 1; m >>= 1) tsum += __shfl_xor(tsum, m, 32);
      lrun[r] = lrun[r] * alpha[r] + tsum;
    }
#pragma unroll
    for (int g = 0; g < 4; ++g)
#pragma unroll
      for (int r = 0; r < 8; ++r) oacc[g][r] *= alpha[r];

    // P (C-layout) -> per-wave LDS -> A-layout fragments.
    // Same-wave DS ops complete in order; no workgroup barrier needed here.
#pragma unroll
    for (int n = 0; n < 4; ++n)
#pragma unroll
      for (int r = 0; r < 8; ++r)
        Ps[wave][(r + 8 * half) * TK + n * 16 + ln] = (bf16_t)s[n][r];

    // out += P (16 x 64 keys) * V; B operand rows = dk from Vs ([dk][key]).
#pragma unroll
    for (int t = 0; t < 2; ++t) {
      v16bf pf = load_frag_k(&Ps[wave][ln * TK + t * 32], half);
#pragma unroll
      for (int g = 0; g < 4; ++g) {
        v16bf vf = load_frag_k(&Vs[sel][(g * 16 + ln) * TK + t * 32], half);
        oacc[g] = wmma_bf16(pf, vf, oacc[g]);
      }
    }
    __syncthreads();  // all reads of buf[sel] done; safe to overwrite next iter
    sel ^= 1;
  }

  // Normalize and store context (bf16).
#pragma unroll
  for (int g = 0; g < 4; ++g)
#pragma unroll
    for (int r = 0; r < 8; ++r) {
      float o = oacc[g][r] / lrun[r];
      int qrow = q0 + wave * 16 + r + 8 * half;
      Ctx[(size_t)(b * S + qrow) * D + h * DKc + g * 16 + ln] = (bf16_t)o;
    }
}

// ----------------------------------------------------------------- launch ----
extern "C" void kernel_launch(void* const* d_in, const int* in_sizes, int n_in,
                              void* d_out, int out_size, void* d_ws, size_t ws_size,
                              hipStream_t stream) {
  constexpr int B = 2, S = 2048, D = 1024, H = 16;
  constexpr size_t BSD = (size_t)B * S * D;  // 4,194,304
  constexpr size_t DD  = (size_t)D * D;      // 1,048,576
  constexpr int NMASK  = B * S * S;          // 8,388,608 mask elements
  constexpr int NWORDS = NMASK / 32;         // 262,144 packed words

  const float* q    = (const float*)d_in[0];
  const float* k    = (const float*)d_in[1];
  const float* v    = (const float*)d_in[2];
  const int*   mask = (const int*)d_in[3];
  const float* wq   = (const float*)d_in[4];
  const float* wk   = (const float*)d_in[5];
  const float* wv   = (const float*)d_in[6];
  const float* wo   = (const float*)d_in[7];

  // Workspace layout: 7*BSD + 4*DD bf16 (~67 MB) + 1 MB packed mask.
  bf16_t* ws  = (bf16_t*)d_ws;
  bf16_t* qb  = ws;
  bf16_t* kb  = qb + BSD;
  bf16_t* vb  = kb + BSD;
  bf16_t* Qp  = vb + BSD;
  bf16_t* Kp  = Qp + BSD;
  bf16_t* Vtg = Kp + BSD;   // V projection stored transposed: [B,H,DK,S]
  bf16_t* Cx  = Vtg + BSD;
  bf16_t* wqb = Cx + BSD;
  bf16_t* wkb = wqb + DD;
  bf16_t* wvb = wkb + DD;
  bf16_t* wob = wvb + DD;
  uint32_t* pmask = (uint32_t*)(wob + DD);

  auto cvt = [&](const float* src, bf16_t* dst, size_t n) {
    int n4 = (int)(n / 4);
    f32_to_bf16_kernel<<<(n4 + 255) / 256, 256, 0, stream>>>(src, dst, n4);
  };
  cvt(q, qb, BSD);
  cvt(k, kb, BSD);
  cvt(v, vb, BSD);
  cvt(wq, wqb, DD);
  cvt(wk, wkb, DD);
  cvt(wv, wvb, DD);
  cvt(wo, wob, DD);
  mask_pack_kernel<<<NMASK / 256, 256, 0, stream>>>(mask, pmask, NWORDS);

  const int M = B * S;                 // 4096
  dim3 ggrid(D / 128, M / 128);        // (8, 32)
  gemm_bf16_nt<bf16_t, false><<<ggrid, 256, 0, stream>>>(qb, wqb, Qp, M, D, D);
  gemm_bf16_nt<bf16_t, false><<<ggrid, 256, 0, stream>>>(kb, wkb, Kp, M, D, D);
  gemm_bf16_nt<bf16_t, true ><<<ggrid, 256, 0, stream>>>(vb, wvb, Vtg, M, D, D);

  dim3 agrid(S / 64, H, B);            // (32, 16, 2)
  flash_attn<<<agrid, 128, 0, stream>>>(Qp, Kp, Vtg, pmask, Cx);

  gemm_bf16_nt<float, false><<<ggrid, 256, 0, stream>>>(Cx, wob, (float*)d_out, M, D, D);
}
